// ALIGNN_PyG_54760833024042
// MI455X (gfx1250) — compile-verified, matrix operand order checked by reading, not run on hardware
//
#include <hip/hip_runtime.h>
#include <hip/hip_bf16.h>
#include <math.h>
#include <stdint.h>

// ---------------------------------------------------------------------------
// ALIGNN forward for MI455X (gfx1250):
//   - WMMA bf16 GEMMs (v_wmma_f32_16x16x32_bf16), f32 accumulate
//   - weight tiles staged to LDS via GLOBAL_LOAD_ASYNC_TO_LDS_B128 (ASYNCcnt)
//   - B fragments hoisted from LDS into registers once per wave
//   - tile-level bounds fast path: unguarded stores except the boundary tile
//   - irregular gather/gate/scatter in fp32 with HW global_atomic_add_f32
// ---------------------------------------------------------------------------

#define HID   64
#define ATOMF 92
#define EBINS 40
#define TBINS 20

typedef __attribute__((ext_vector_type(16))) __bf16 v16bf;
typedef __attribute__((ext_vector_type(8)))  __bf16 v8bf;
typedef __attribute__((ext_vector_type(8)))  float  v8f;

union FragBF { v16bf v; v8bf h[2]; };

// Async copy of 16 bytes global -> LDS, tracked by ASYNCcnt.
// VDST = VGPR holding LDS byte address, VADDR = 64-bit global address.
__device__ __forceinline__ void async_b128_to_lds(void* lds_ptr, const void* gptr) {
  uint32_t lds_addr = (uint32_t)(uintptr_t)lds_ptr;  // LDS aperture: addr[31:0]
  asm volatile("global_load_async_to_lds_b128 %0, %1, off"
               :: "v"(lds_addr), "v"(gptr) : "memory");
}
__device__ __forceinline__ void wait_async0() {
  asm volatile("s_wait_asynccnt 0" ::: "memory");
}

// Load a 16x32 bf16 fragment (A or B operand) from a row-major tile.
// ISA 16-bit layout: lanes 0-15 hold M=lane, K={0..7,16..23};
// lanes 16-31 hold M=lane-16, K={8..15,24..31}.  Two 16B loads per lane.
__device__ __forceinline__ v16bf load_frag16x32(const __bf16* __restrict__ tile,
                                                int ld, int k0, int lane) {
  const int r  = lane & 15;
  const int hi = (lane >> 4) & 1;
  const __bf16* p = tile + (size_t)r * ld + k0 + hi * 8;
  FragBF f;
  f.h[0] = *(const v8bf*)(p);        // K = k0 + hi*8      .. +7
  f.h[1] = *(const v8bf*)(p + 16);   // K = k0 + 16 + hi*8 .. +7
  return f.v;
}

__device__ __forceinline__ float siluf(float x) { return x / (1.f + __expf(-x)); }

// ---------------------------------------------------------------------------
// Generic GEMM: out[M x 64] = A[M x K](bf16) @ Wt^T + bias, optional LN+SiLU.
// Templated on K so the B fragments (4 N-tiles x K/32 K-steps) are preloaded
// into registers from the async-staged LDS weight tile; the MMA sequence then
// depends only on the A-stream global loads.
// mode 0: bias only.  mode 1: bias + LayerNorm(64) + SiLU.
// ---------------------------------------------------------------------------
template <int K>
__global__ void __launch_bounds__(256) gemm_ln_kernel(
    const __bf16* __restrict__ A,
    const __bf16* __restrict__ Wt,                   // Wt: [64 rows(N) x K]
    const float* __restrict__ bias,
    const float* __restrict__ ln_g, const float* __restrict__ ln_b,
    float* __restrict__ out_f32, __bf16* __restrict__ out_bf16,
    int M, int mode)
{
  constexpr int KS = K / 32;
  __shared__ __bf16 swt[64 * K];

  // Cooperative async stage of the weight tile into LDS.
  {
    constexpr int chunks = (64 * K) / 8;             // 16B chunks
    for (int i = threadIdx.x; i < chunks; i += 256)
      async_b128_to_lds(&swt[i * 8], Wt + (size_t)i * 8);
    wait_async0();
    __syncthreads();
  }

  const int lane = threadIdx.x & 31;
  const int wave = threadIdx.x >> 5;
  const int row0 = (blockIdx.x * 8 + wave) * 16;

  // Hoist all B fragments into registers (one ds burst, one wait).
  v16bf bf[4][KS];
#pragma unroll
  for (int j = 0; j < 4; ++j)
#pragma unroll
    for (int ks = 0; ks < KS; ++ks)
      bf[j][ks] = load_frag16x32(swt + (size_t)(j * 16) * K, K, ks * 32, lane);

  v8f acc[4] = {};
  const __bf16* Arow = A + (size_t)row0 * K;
#pragma unroll
  for (int ks = 0; ks < KS; ++ks) {
    v16bf a = load_frag16x32(Arow, K, ks * 32, lane);
#pragma unroll
    for (int j = 0; j < 4; ++j)
      acc[j] = __builtin_amdgcn_wmma_f32_16x16x32_bf16(
          false, a, false, bf[j][ks], (short)0, acc[j], false, false);
  }

  const int nlo = lane & 15;
  const int hi  = (lane >> 4) & 1;

#pragma unroll
  for (int j = 0; j < 4; ++j) {
    float bv = bias ? bias[j * 16 + nlo] : 0.f;
#pragma unroll
    for (int r = 0; r < 8; ++r) acc[j][r] += bv;
  }

  if (mode == 1) {
    // Row m = row0 + hi*8 + r lives in one VGPR slot across 16 lanes x 4 tiles.
#pragma unroll
    for (int r = 0; r < 8; ++r) {
      float s = acc[0][r] + acc[1][r] + acc[2][r] + acc[3][r];
      float q = acc[0][r]*acc[0][r] + acc[1][r]*acc[1][r]
              + acc[2][r]*acc[2][r] + acc[3][r]*acc[3][r];
#pragma unroll
      for (int off = 1; off < 16; off <<= 1) {   // stays inside each 16-lane half
        s += __shfl_xor(s, off, 32);
        q += __shfl_xor(q, off, 32);
      }
      float mean = s * (1.f / 64.f);
      float var  = q * (1.f / 64.f) - mean * mean;
      float inv  = rsqrtf(var + 1e-5f);
#pragma unroll
      for (int j = 0; j < 4; ++j) {
        int n = j * 16 + nlo;
        float y = (acc[j][r] - mean) * inv * ln_g[n] + ln_b[n];
        acc[j][r] = siluf(y);
      }
    }
  }

  // Tile-level bounds fast path: only the boundary tile takes guarded stores.
  if (row0 + 16 <= M) {
#pragma unroll
    for (int j = 0; j < 4; ++j)
#pragma unroll
      for (int r = 0; r < 8; ++r) {
        size_t o = (size_t)(row0 + hi * 8 + r) * HID + j * 16 + nlo;
        if (out_f32)  out_f32[o]  = acc[j][r];
        if (out_bf16) out_bf16[o] = (__bf16)acc[j][r];
      }
  } else {
#pragma unroll
    for (int j = 0; j < 4; ++j)
#pragma unroll
      for (int r = 0; r < 8; ++r) {
        int mrow = row0 + hi * 8 + r;
        if (mrow < M) {
          size_t o = (size_t)mrow * HID + j * 16 + nlo;
          if (out_f32)  out_f32[o]  = acc[j][r];
          if (out_bf16) out_bf16[o] = (__bf16)acc[j][r];
        }
      }
  }
}

// ---------------------------------------------------------------------------
// Fused 4-way projection: gi,gj,uj,sux = x@{sg,dg,du,su} + b.
// The two A fragments are loaded ONCE per wave (4 global b128 total) and
// reused by all 32 WMMAs; each weight's 8 B fragments come from the
// async-staged 32 KB LDS tile, and its accumulators retire before the next
// weight starts (stores overlap the next ds burst).
// ---------------------------------------------------------------------------
__global__ void __launch_bounds__(256) gemm4_kernel(
    const __bf16* __restrict__ A,
    const __bf16* __restrict__ W0, const __bf16* __restrict__ W1,
    const __bf16* __restrict__ W2, const __bf16* __restrict__ W3,
    const float* __restrict__ b0, const float* __restrict__ b1,
    const float* __restrict__ b2, const float* __restrict__ b3,
    float* __restrict__ o0, float* __restrict__ o1,
    float* __restrict__ o2, float* __restrict__ o3,
    int M)
{
  __shared__ __bf16 swt[4 * 64 * 64];                // 32 KB

  const __bf16* Wts[4] = {W0, W1, W2, W3};
  const float*  Bs[4]  = {b0, b1, b2, b3};
  float*        Os[4]  = {o0, o1, o2, o3};

  {
    // 4 * 4096 bf16 = 2048 x 16B chunks, 8 per thread.
    for (int i = threadIdx.x; i < 2048; i += 256) {
      int w = i >> 9, c = i & 511;
      async_b128_to_lds(&swt[w * 4096 + c * 8], Wts[w] + (size_t)c * 8);
    }
    wait_async0();
    __syncthreads();
  }

  const int lane = threadIdx.x & 31;
  const int wave = threadIdx.x >> 5;
  const int row0 = (blockIdx.x * 8 + wave) * 16;
  const int nlo  = lane & 15;
  const int hi   = (lane >> 4) & 1;
  const bool full = (row0 + 16 <= M);

  const __bf16* Arow = A + (size_t)row0 * HID;
  v16bf a0 = load_frag16x32(Arow, HID, 0, lane);
  v16bf a1 = load_frag16x32(Arow, HID, 32, lane);

#pragma unroll
  for (int w = 0; w < 4; ++w) {
    const __bf16* wt = swt + w * 4096;
    v8f acc[4] = {};
#pragma unroll
    for (int j = 0; j < 4; ++j) {
      v16bf bA = load_frag16x32(wt + (size_t)(j * 16) * HID, HID, 0, lane);
      v16bf bB = load_frag16x32(wt + (size_t)(j * 16) * HID, HID, 32, lane);
      acc[j] = __builtin_amdgcn_wmma_f32_16x16x32_bf16(
          false, a0, false, bA, (short)0, acc[j], false, false);
      acc[j] = __builtin_amdgcn_wmma_f32_16x16x32_bf16(
          false, a1, false, bB, (short)0, acc[j], false, false);
    }
    if (full) {
#pragma unroll
      for (int j = 0; j < 4; ++j) {
        float bv = Bs[w][j * 16 + nlo];
#pragma unroll
        for (int r = 0; r < 8; ++r)
          Os[w][(size_t)(row0 + hi * 8 + r) * HID + j * 16 + nlo] =
              acc[j][r] + bv;
      }
    } else {
#pragma unroll
      for (int j = 0; j < 4; ++j) {
        float bv = Bs[w][j * 16 + nlo];
#pragma unroll
        for (int r = 0; r < 8; ++r) {
          int mrow = row0 + hi * 8 + r;
          if (mrow < M)
            Os[w][(size_t)mrow * HID + j * 16 + nlo] = acc[j][r] + bv;
        }
      }
    }
  }
}

// ---------------------------------------------------------------------------
// Per-edge gate + message + scatter-add (fp32 HW atomics).
// One wave per edge, 2 columns per lane (float2).
// ---------------------------------------------------------------------------
__global__ void __launch_bounds__(256) egc_scatter_kernel(
    const float* __restrict__ gi, const float* __restrict__ gj,
    const float* __restrict__ uj, const float* __restrict__ ege,
    const int* __restrict__ src, const int* __restrict__ dst,
    float* __restrict__ aggr, int m)
{
  int t = blockIdx.x * 8 + (threadIdx.x >> 5);
  if (t >= m) return;
  int lane = threadIdx.x & 31;
  int c = lane * 2;
  int s = src[t], d = dst[t];
  float2 gi2 = *(const float2*)(gi + (size_t)d * HID + c);
  float2 gj2 = *(const float2*)(gj + (size_t)s * HID + c);
  float2 eg2 = *(const float2*)(ege + (size_t)t * HID + c);
  float2 uj2 = *(const float2*)(uj + (size_t)s * HID + c);
  float g0 = siluf(gi2.x + gj2.x + eg2.x);
  float g1 = siluf(gi2.y + gj2.y + eg2.y);
  unsafeAtomicAdd(&aggr[(size_t)d * HID + c],     g0 * uj2.x);
  unsafeAtomicAdd(&aggr[(size_t)d * HID + c + 1], g1 * uj2.y);
}

// out = silu(LN(sux + aggr)) + x_in   (one wave per row, 2 cols/lane)
__global__ void __launch_bounds__(256) egc_update_kernel(
    const float* __restrict__ sux, const float* __restrict__ aggr,
    const float* __restrict__ x_in,
    const float* __restrict__ ln_g, const float* __restrict__ ln_b,
    float* __restrict__ out_f32, __bf16* __restrict__ out_bf16, int n)
{
  int row = blockIdx.x * 8 + (threadIdx.x >> 5);
  if (row >= n) return;
  int lane = threadIdx.x & 31;
  int c = lane * 2;
  size_t o = (size_t)row * HID + c;
  float v0 = sux[o]     + aggr[o];
  float v1 = sux[o + 1] + aggr[o + 1];
  float s = v0 + v1, q = v0 * v0 + v1 * v1;
#pragma unroll
  for (int off = 1; off < 32; off <<= 1) {
    s += __shfl_xor(s, off, 32);
    q += __shfl_xor(q, off, 32);
  }
  float mean = s * (1.f / 64.f);
  float var  = q * (1.f / 64.f) - mean * mean;
  float inv  = rsqrtf(var + 1e-5f);
  float y0 = siluf((v0 - mean) * inv * ln_g[c]     + ln_b[c])     + x_in[o];
  float y1 = siluf((v1 - mean) * inv * ln_g[c + 1] + ln_b[c + 1]) + x_in[o + 1];
  out_f32[o] = y0;     out_f32[o + 1] = y1;
  out_bf16[o] = (__bf16)y0; out_bf16[o + 1] = (__bf16)y1;
}

// RBF expansion -> zero-padded bf16 feature matrix [npad x kpad]
__global__ void rbf_kernel(const float* __restrict__ d, __bf16* __restrict__ out,
                           int n, int npad, int bins, int kpad,
                           float vmin, float vmax)
{
  size_t i = (size_t)blockIdx.x * blockDim.x + threadIdx.x;
  size_t total = (size_t)npad * kpad;
  if (i >= total) return;
  int row = (int)(i / kpad), col = (int)(i % kpad);
  float val = 0.f;
  if (row < n && col < bins) {
    float step = (vmax - vmin) / (bins - 1);
    float ctr = vmin + col * step;
    float g = 1.f / (step * step);
    float x = d[row] - ctr;
    val = __expf(-g * x * x);
  }
  out[i] = (__bf16)val;
}

// fp32 [n x k] -> zero-padded bf16 [npad x kpad]
__global__ void castpad_kernel(const float* __restrict__ in, __bf16* __restrict__ out,
                               int n, int npad, int k, int kpad)
{
  size_t i = (size_t)blockIdx.x * blockDim.x + threadIdx.x;
  size_t total = (size_t)npad * kpad;
  if (i >= total) return;
  int row = (int)(i / kpad), col = (int)(i % kpad);
  out[i] = (__bf16)((row < n && col < k) ? in[(size_t)row * k + col] : 0.f);
}

// W[K x 64] fp32 -> Wt[64 x Kpad] bf16 (transposed + zero-padded)
__global__ void prep_wt_kernel(const float* __restrict__ w, __bf16* __restrict__ wt,
                               int K, int kpad)
{
  int i = blockIdx.x * blockDim.x + threadIdx.x;
  if (i >= 64 * kpad) return;
  int nrow = i / kpad, kcol = i % kpad;
  wt[i] = (__bf16)((kcol < K) ? w[(size_t)kcol * 64 + nrow] : 0.f);
}

__global__ void __launch_bounds__(256) pool_kernel(
    const float* __restrict__ x, const int* __restrict__ batch,
    float* __restrict__ psum, float* __restrict__ pcnt, int n)
{
  int row = blockIdx.x * 8 + (threadIdx.x >> 5);
  if (row >= n) return;
  int lane = threadIdx.x & 31;
  int c = lane * 2;
  int g = batch[row];
  unsafeAtomicAdd(&psum[(size_t)g * HID + c],     x[(size_t)row * HID + c]);
  unsafeAtomicAdd(&psum[(size_t)g * HID + c + 1], x[(size_t)row * HID + c + 1]);
  if (lane == 0) unsafeAtomicAdd(&pcnt[g], 1.f);
}

// mean-pool -> silu(fc) -> out (one 64-thread block per graph)
__global__ void __launch_bounds__(64) head_kernel(
    const float* __restrict__ psum, const float* __restrict__ pcnt,
    const float* __restrict__ fc_w, const float* __restrict__ fc_b,
    const float* __restrict__ out_w, const float* __restrict__ out_b,
    float* __restrict__ out)
{
  __shared__ float pool[64];
  __shared__ float h[64];
  int g = blockIdx.x, t = threadIdx.x;
  float cnt = fmaxf(pcnt[g], 1.f);
  pool[t] = psum[(size_t)g * HID + t] / cnt;
  __syncthreads();
  float acc = fc_b[t];
  for (int k = 0; k < 64; ++k) acc += pool[k] * fc_w[(size_t)k * 64 + t];
  h[t] = siluf(acc) * out_w[t];
  __syncthreads();
  for (int off = 32; off > 0; off >>= 1) {
    if (t < off) h[t] += h[t + off];
    __syncthreads();
  }
  if (t == 0) out[g] = h[0] + out_b[0];
}

// ---------------------------------------------------------------------------
// Host orchestration
// ---------------------------------------------------------------------------
struct Egc {  // leaf order inside one _egc_params dict (sorted keys)
  const float *dg_b, *dg_w, *du_b, *du_w, *eg_b, *eg_w,
              *ln_b, *ln_g, *sg_b, *sg_w, *su_b, *su_w;
};
struct Emb { const float *b, *ln_b, *ln_g, *w; };  // sorted: b, ln_b, ln_g, w
struct EgcWts { const __bf16 *sg, *dg, *du, *su, *eg; };

static Egc get_egc(void* const* din, int base) {
  Egc e;
  e.dg_b = (const float*)din[base + 0];  e.dg_w = (const float*)din[base + 1];
  e.du_b = (const float*)din[base + 2];  e.du_w = (const float*)din[base + 3];
  e.eg_b = (const float*)din[base + 4];  e.eg_w = (const float*)din[base + 5];
  e.ln_b = (const float*)din[base + 6];  e.ln_g = (const float*)din[base + 7];
  e.sg_b = (const float*)din[base + 8];  e.sg_w = (const float*)din[base + 9];
  e.su_b = (const float*)din[base + 10]; e.su_w = (const float*)din[base + 11];
  return e;
}
static Emb get_emb(void* const* din, int base) {
  Emb m;
  m.b    = (const float*)din[base + 0];
  m.ln_b = (const float*)din[base + 1];
  m.ln_g = (const float*)din[base + 2];
  m.w    = (const float*)din[base + 3];
  return m;
}

extern "C" void kernel_launch(void* const* d_in, const int* in_sizes, int n_in,
                              void* d_out, int out_size, void* d_ws, size_t ws_size,
                              hipStream_t stream)
{
  const float* x_atom    = (const float*)d_in[0];
  const float* edge_dist = (const float*)d_in[1];
  const float* angle_cos = (const float*)d_in[2];
  const int*   edge_index = (const int*)d_in[n_in - 4];
  const int*   lg_index   = (const int*)d_in[n_in - 3];
  const int*   batch      = (const int*)d_in[n_in - 2];

  const int N = in_sizes[0] / ATOMF;
  const int E = in_sizes[1];
  const int T = in_sizes[2];
  const int G = out_size;

  auto pad128 = [](int n) { return (n + 127) & ~127; };
  const int Npad = pad128(N), Epad = pad128(E), Tpad = pad128(T);

  // ---- params pytree (sorted-key flatten), starting at d_in[3] ----
  const int P = 3;
  Egc al_edge[4], al_node[4], gcn[4];
  EgcWts al_edge_w[4], al_node_w[4], gcn_w[4];
  for (int i = 0; i < 4; ++i) {
    al_edge[i] = get_egc(d_in, P + i * 24);
    al_node[i] = get_egc(d_in, P + i * 24 + 12);
  }
  Emb angle_emb = get_emb(d_in, P + 96);
  Emb atom_emb  = get_emb(d_in, P + 100);
  Emb edge_emb  = get_emb(d_in, P + 104);
  const float* fc_b  = (const float*)d_in[P + 108];
  const float* fc_w  = (const float*)d_in[P + 109];
  for (int i = 0; i < 4; ++i) gcn[i] = get_egc(d_in, P + 110 + i * 12);
  const float* out_b = (const float*)d_in[P + 158];
  const float* out_w = (const float*)d_in[P + 159];

  // ---- workspace carve-out ----
  char* wsb = (char*)d_ws;
  size_t cur = 0;
  auto alloc = [&](size_t bytes) -> void* {
    void* p = wsb + cur;
    cur = (cur + bytes + 255) & ~(size_t)255;
    return p;
  };
  (void)ws_size;

  auto prep = [&](const float* w, int K, int Kpad) -> const __bf16* {
    __bf16* wt = (__bf16*)alloc((size_t)64 * Kpad * sizeof(__bf16));
    int tot = 64 * Kpad;
    prep_wt_kernel<<<(tot + 255) / 256, 256, 0, stream>>>(w, wt, K, Kpad);
    return wt;
  };
  auto prep_egc = [&](const Egc& p) -> EgcWts {
    EgcWts w;
    w.sg = prep(p.sg_w, HID, HID);  w.dg = prep(p.dg_w, HID, HID);
    w.du = prep(p.du_w, HID, HID);  w.su = prep(p.su_w, HID, HID);
    w.eg = prep(p.eg_w, HID, HID);
    return w;
  };

  const __bf16* atomWt  = prep(atom_emb.w, ATOMF, 96);
  const __bf16* edgeWt  = prep(edge_emb.w, EBINS, 64);
  const __bf16* angleWt = prep(angle_emb.w, TBINS, 32);
  for (int i = 0; i < 4; ++i) {
    al_edge_w[i] = prep_egc(al_edge[i]);
    al_node_w[i] = prep_egc(al_node[i]);
    gcn_w[i]     = prep_egc(gcn[i]);
  }

  // activations (padded rows)
  __bf16* featA  = (__bf16*)alloc((size_t)Npad * 96 * 2);
  __bf16* featE  = (__bf16*)alloc((size_t)Epad * 64 * 2);
  __bf16* featT  = (__bf16*)alloc((size_t)Tpad * 32 * 2);
  float*  xf[2]  = {(float*)alloc((size_t)Npad * HID * 4),
                    (float*)alloc((size_t)Npad * HID * 4)};
  __bf16* xb[2]  = {(__bf16*)alloc((size_t)Npad * HID * 2),
                    (__bf16*)alloc((size_t)Npad * HID * 2)};
  float*  ef[2]  = {(float*)alloc((size_t)Epad * HID * 4),
                    (float*)alloc((size_t)Epad * HID * 4)};
  __bf16* eb[2]  = {(__bf16*)alloc((size_t)Epad * HID * 2),
                    (__bf16*)alloc((size_t)Epad * HID * 2)};
  __bf16* ab     = (__bf16*)alloc((size_t)Tpad * HID * 2);
  float*  gi     = (float*)alloc((size_t)Epad * HID * 4);
  float*  gj     = (float*)alloc((size_t)Epad * HID * 4);
  float*  uj     = (float*)alloc((size_t)Epad * HID * 4);
  float*  sux    = (float*)alloc((size_t)Epad * HID * 4);
  float*  ege    = (float*)alloc((size_t)Tpad * HID * 4);
  float*  aggr   = (float*)alloc((size_t)Epad * HID * 4);
  float*  psum   = (float*)alloc((size_t)G * HID * 4);
  float*  pcnt   = (float*)alloc((size_t)G * 4);

  // ---- input featurization ----
  {
    size_t tot = (size_t)Npad * 96;
    castpad_kernel<<<(unsigned)((tot + 255) / 256), 256, 0, stream>>>(
        x_atom, featA, N, Npad, ATOMF, 96);
    tot = (size_t)Epad * 64;
    rbf_kernel<<<(unsigned)((tot + 255) / 256), 256, 0, stream>>>(
        edge_dist, featE, E, Epad, EBINS, 64, 0.f, 10.f);
    tot = (size_t)Tpad * 32;
    rbf_kernel<<<(unsigned)((tot + 255) / 256), 256, 0, stream>>>(
        angle_cos, featT, T, Tpad, TBINS, 32, -1.f, 1.f);
  }

  // ---- embeddings (GEMM + LN + SiLU via WMMA) ----
  gemm_ln_kernel<96><<<Npad / 128, 256, 0, stream>>>(
      featA, atomWt, atom_emb.b, atom_emb.ln_g, atom_emb.ln_b,
      xf[0], xb[0], N, 1);
  gemm_ln_kernel<64><<<Epad / 128, 256, 0, stream>>>(
      featE, edgeWt, edge_emb.b, edge_emb.ln_g, edge_emb.ln_b,
      ef[0], eb[0], E, 1);
  gemm_ln_kernel<32><<<Tpad / 128, 256, 0, stream>>>(
      featT, angleWt, angle_emb.b, angle_emb.ln_g, angle_emb.ln_b,
      nullptr, ab, T, 1);

  // ---- one edge-gated graph convolution ----
  auto run_egc = [&](const Egc& p, const EgcWts& w,
                     const __bf16* xin_b, const float* xin_f,
                     float* xout_f, __bf16* xout_b,
                     const int* src, const int* dst, int m, int mpad,
                     const __bf16* eattr_b, int n, int npad) {
    gemm4_kernel<<<npad / 128, 256, 0, stream>>>(
        xin_b, w.sg, w.dg, w.du, w.su,
        p.sg_b, p.dg_b, p.du_b, p.su_b, gi, gj, uj, sux, n);
    gemm_ln_kernel<64><<<mpad / 128, 256, 0, stream>>>(
        eattr_b, w.eg, p.eg_b, nullptr, nullptr, ege, nullptr, m, 0);
    hipMemsetAsync(aggr, 0, (size_t)n * HID * sizeof(float), stream);
    egc_scatter_kernel<<<(m + 7) / 8, 256, 0, stream>>>(
        gi, gj, uj, ege, src, dst, aggr, m);
    egc_update_kernel<<<(n + 7) / 8, 256, 0, stream>>>(
        sux, aggr, xin_f, p.ln_g, p.ln_b, xout_f, xout_b, n);
  };

  int xc = 0, ec = 0;
  for (int i = 0; i < 4; ++i) {
    // edge update on line graph: e' = EGC(e, lg_index, a)
    run_egc(al_edge[i], al_edge_w[i], eb[ec], ef[ec], ef[1 - ec], eb[1 - ec],
            lg_index, lg_index + T, T, Tpad, ab, E, Epad);
    ec = 1 - ec;
    // node update on bond graph: x' = EGC(x, edge_index, e')
    run_egc(al_node[i], al_node_w[i], xb[xc], xf[xc], xf[1 - xc], xb[1 - xc],
            edge_index, edge_index + E, E, Epad, eb[ec], N, Npad);
    xc = 1 - xc;
  }
  for (int i = 0; i < 4; ++i) {
    run_egc(gcn[i], gcn_w[i], xb[xc], xf[xc], xf[1 - xc], xb[1 - xc],
            edge_index, edge_index + E, E, Epad, eb[ec], N, Npad);
    xc = 1 - xc;
  }

  // ---- readout ----
  hipMemsetAsync(psum, 0, (size_t)G * HID * sizeof(float), stream);
  hipMemsetAsync(pcnt, 0, (size_t)G * sizeof(float), stream);
  pool_kernel<<<(N + 7) / 8, 256, 0, stream>>>(xf[xc], batch, psum, pcnt, N);
  head_kernel<<<G, 64, 0, stream>>>(psum, pcnt, fc_w, fc_b, out_w, out_b,
                                    (float*)d_out);
}
